// LSTM_80092550136430
// MI455X (gfx1250) — matrix operand order repaired and checked by reading, not compiled, not run on hardware
//
#include <hip/hip_runtime.h>
#include <math.h>

// ---------------------------------------------------------------------------
// LSTM (LAGS=64, B=8192, X=128, H=256) via bf16 WMMA on gfx1250.
//   pack_weights : Wall[384][1024] -> fragment-major bf16 B operands (768 KB ws)
//   lstm_main    : 64 batch rows / block (128 blocks, 8 waves).
//     - wave w owns h-cols {p*128 + w*16 + 0..15} for p=0,1, all 4 gates,
//       4 M-tiles => per pass acc[4][4] v8f, persistent C-state cst[2][4] v8f
//       kept in the WMMA C/D register layout (no memory for the recurrence).
//     - A operand ([H|x] bf16) double-buffered in LDS: one barrier per step.
//     - kt loop kept rolled (#pragma unroll 1) so weight loads cannot be
//       hoisted out of the time loop (avoids scratch spills).
//     - branch-free sigmoid/tanh on v_exp_f32/v_rcp_f32 (TRANS ops co-execute
//       with WMMA); libm tanhf's divergent EXEC-mask path removed.
// ---------------------------------------------------------------------------

#define LAGS   64
#define BATCH  8192
#define X_DIM  128
#define HID    256
#define Y_DIM  64
#define KTOT   384          // HID + X_DIM
#define NTOT   1024         // 4 * HID
#define NKT    12           // KTOT / 32
#define TILEB  64           // batch rows per block
#define ROWSTRIDE 392       // 384 + 8 halfs pad (row stride = 784 B, 16B aligned)

typedef __attribute__((ext_vector_type(16))) __bf16 v16bf;
typedef __attribute__((ext_vector_type(8)))  float  v8f;

union FragBF {
    uint4 u[2];
    v16bf v;
};

__device__ __forceinline__ unsigned short f2bf(float f) {
    unsigned int u = __float_as_uint(f);
    u += 0x7fffu + ((u >> 16) & 1u);          // round-to-nearest-even
    return (unsigned short)(u >> 16);
}
__device__ __forceinline__ float bf2f(unsigned short h) {
    return __uint_as_float(((unsigned int)h) << 16);
}
// branch-free fast activations: v_exp_f32 + v_rcp_f32 only
__device__ __forceinline__ float fast_sigm(float x) {
    return __builtin_amdgcn_rcpf(1.0f + __expf(-x));
}
__device__ __forceinline__ float fast_tanh(float x) {
    // tanh(x) = 2*sigmoid(2x) - 1
    return 2.0f * __builtin_amdgcn_rcpf(1.0f + __expf(-2.0f * x)) - 1.0f;
}

// ---------------------------------------------------------------------------
// Pack combined weight matrix Wall[384][1024] into per-lane WMMA B fragments.
// Fragment id = nt*NKT + kt ; element addr = (fragid*32 + lane)*16 + j
//   lane <  16 : element j = Wall[kt*32 +      j][nt*16 + lane]
//   lane >= 16 : element j = Wall[kt*32 + 16 + j][nt*16 + lane - 16]
// ---------------------------------------------------------------------------
__global__ void __launch_bounds__(256)
pack_weights(const float* __restrict__ Wf, const float* __restrict__ Wi,
             const float* __restrict__ Wc, const float* __restrict__ Wo,
             const float* __restrict__ Uf, const float* __restrict__ Ui,
             const float* __restrict__ Uc, const float* __restrict__ Uo,
             unsigned short* __restrict__ packed)
{
    int id = blockIdx.x * blockDim.x + threadIdx.x;   // 0 .. 384*1024-1
    int j    = id & 15;
    int lane = (id >> 4) & 31;
    int tile = id >> 9;                               // nt*NKT + kt
    int kt = tile % NKT;
    int nt = tile / NKT;

    int k = kt * 32 + ((lane >= 16) ? 16 : 0) + j;    // 0..383
    int n = nt * 16 + (lane & 15);                    // 0..1023
    int g   = n >> 8;                                 // gate: f,i,C,o
    int col = n & 255;

    const float* W[4] = {Wf, Wi, Wc, Wo};
    const float* U[4] = {Uf, Ui, Uc, Uo};
    float val = (k < HID) ? W[g][k * HID + col]
                          : U[g][(k - HID) * HID + col];
    packed[id] = f2bf(val);
}

// ---------------------------------------------------------------------------
// Main fused LSTM kernel. grid = BATCH/TILEB blocks, 256 threads (8 waves).
// ---------------------------------------------------------------------------
__global__ void __launch_bounds__(256)
lstm_main(const float* __restrict__ X,
          const unsigned short* __restrict__ packed,
          const float* __restrict__ b_f, const float* __restrict__ b_i,
          const float* __restrict__ b_c, const float* __restrict__ b_o,
          const float* __restrict__ V,   const float* __restrict__ bias_out,
          const float* __restrict__ fc1_w, const float* __restrict__ fc1_b,
          const float* __restrict__ fc2_w, const float* __restrict__ fc2_b,
          float* __restrict__ out)
{
    // double-buffered A operand: [buf][row][k] bf16, 2 * 64 * 392 * 2B = 98 KB
    __shared__ __align__(16) unsigned short Abuf[2][TILEB * ROWSTRIDE];

    const int tid  = threadIdx.x;
    const int lane = tid & 31;
    const int w    = tid >> 5;                 // wave id 0..7
    const int b0   = blockIdx.x * TILEB;       // batch-row base

    // ---- zero H region of buf0 (cols 0..255, all 64 rows) -----------------
    {
        int r  = tid >> 2;                     // 0..63
        int c0 = (tid & 3) * 64;               // 64 halfs per thread
        uint4 z = {0u, 0u, 0u, 0u};
        #pragma unroll
        for (int k = 0; k < 8; ++k)
            *(uint4*)&Abuf[0][r * ROWSTRIDE + c0 + k * 8] = z;
    }
    // ---- x(0) into buf0 (cols 256..383) -----------------------------------
    {
        int r  = tid >> 2;
        int c0 = (tid & 3) * 32;               // 32 floats per thread
        const float* src = X + ((size_t)b0 + r) * X_DIM + c0;
        #pragma unroll
        for (int k = 0; k < 4; ++k) {
            float4 f0 = *(const float4*)(src + k * 8);
            float4 f1 = *(const float4*)(src + k * 8 + 4);
            union { unsigned short h[8]; uint4 u; } pk;
            pk.h[0] = f2bf(f0.x); pk.h[1] = f2bf(f0.y);
            pk.h[2] = f2bf(f0.z); pk.h[3] = f2bf(f0.w);
            pk.h[4] = f2bf(f1.x); pk.h[5] = f2bf(f1.y);
            pk.h[6] = f2bf(f1.z); pk.h[7] = f2bf(f1.w);
            *(uint4*)&Abuf[0][r * ROWSTRIDE + HID + c0 + k * 8] = pk.u;
        }
    }

    // ---- per-lane biases: bias[p][gate] for h-col p*128 + w*16 + lane%16 --
    float bias[2][4];
    {
        const float* bp[4] = {b_f, b_i, b_c, b_o};
        #pragma unroll
        for (int p = 0; p < 2; ++p) {
            int col = p * 128 + w * 16 + (lane & 15);
            #pragma unroll
            for (int g = 0; g < 4; ++g) bias[p][g] = bp[g][col];
        }
    }

    v8f cst[2][4] = {};                        // C-state [pass][m], C/D layout

    const int rowb  = ((lane >> 4) & 1) * 8;   // C/D row sub-offset
    const int ahalf = ((lane >> 4) & 1) * 16;  // A K-half select (halfs)
    int hcol[2];
    hcol[0] = w * 16 + (lane & 15);
    hcol[1] = 128 + w * 16 + (lane & 15);

    __syncthreads();

    #pragma unroll 1
    for (int t = 0; t < LAGS; ++t) {
        const int buf = t & 1;
        const int nxt = buf ^ 1;
        const unsigned short* __restrict__ Ain = &Abuf[buf][0];
        unsigned short* __restrict__ Aout = &Abuf[nxt][0];

        // ---- two N-passes: pass p covers h-cols [p*128, p*128+128) --------
        #pragma unroll
        for (int p = 0; p < 2; ++p) {
            v8f acc[4][4] = {};                         // [gate][m]
            const unsigned short* bbase[4];
            #pragma unroll
            for (int g = 0; g < 4; ++g)
                bbase[g] = packed + (size_t)((g * 16 + p * 8 + w) * NKT) * 512
                                  + lane * 16;
            const unsigned short* abase =
                Ain + (lane & 15) * ROWSTRIDE + ahalf;

            #pragma unroll 1                 // keep rolled: no LICM weight hoist
            for (int kt = 0; kt < NKT; ++kt) {
                FragBF fb[4], fa[4];
                #pragma unroll
                for (int g = 0; g < 4; ++g) {
                    const unsigned short* bp = bbase[g] + kt * 512;
                    fb[g].u[0] = *(const uint4*)(bp);
                    fb[g].u[1] = *(const uint4*)(bp + 8);
                }
                #pragma unroll
                for (int m = 0; m < 4; ++m) {
                    const unsigned short* ap =
                        abase + m * 16 * ROWSTRIDE + kt * 32;
                    fa[m].u[0] = *(const uint4*)(ap);
                    fa[m].u[1] = *(const uint4*)(ap + 8);
                }
                #pragma unroll
                for (int g = 0; g < 4; ++g) {
                    #pragma unroll
                    for (int m = 0; m < 4; ++m) {
                        acc[g][m] = __builtin_amdgcn_wmma_f32_16x16x32_bf16(
                            false, fa[m].v, false, fb[g].v,
                            (short)0, acc[g][m], false, false);
                    }
                }
            }

            // ---- gate nonlinearities + state update (registers only) ------
            #pragma unroll
            for (int m = 0; m < 4; ++m) {
                #pragma unroll
                for (int v = 0; v < 8; ++v) {
                    float F  = fast_sigm(acc[0][m][v] + bias[p][0]);
                    float I  = fast_sigm(acc[1][m][v] + bias[p][1]);
                    float Cn = fast_tanh(acc[2][m][v] + bias[p][2]);
                    float O  = fast_sigm(acc[3][m][v] + bias[p][3]);
                    float c  = F * cst[p][m][v] + I * Cn;
                    cst[p][m][v] = c;
                    float h  = O * fast_tanh(c);
                    Aout[(m * 16 + rowb + v) * ROWSTRIDE + hcol[p]] = f2bf(h);
                }
            }
        }

        // ---- stream x(t+1) into next buffer -------------------------------
        if (t + 1 < LAGS) {
            int r  = tid >> 2;
            int c0 = (tid & 3) * 32;
            const float* src =
                X + (((size_t)(t + 1) * BATCH) + b0 + r) * X_DIM + c0;
            #pragma unroll
            for (int k = 0; k < 4; ++k) {
                float4 f0 = *(const float4*)(src + k * 8);
                float4 f1 = *(const float4*)(src + k * 8 + 4);
                union { unsigned short h[8]; uint4 u; } pk;
                pk.h[0] = f2bf(f0.x); pk.h[1] = f2bf(f0.y);
                pk.h[2] = f2bf(f0.z); pk.h[3] = f2bf(f0.w);
                pk.h[4] = f2bf(f1.x); pk.h[5] = f2bf(f1.y);
                pk.h[6] = f2bf(f1.z); pk.h[7] = f2bf(f1.w);
                *(uint4*)&Aout[r * ROWSTRIDE + HID + c0 + k * 8] = pk.u;
            }
            if (t + 2 < LAGS)   // gfx1250 global_prefetch_b8
                __builtin_prefetch(src + (size_t)BATCH * X_DIM, 0, 1);
        }
        __syncthreads();    // publish H(t+1), x(t+1) before next GEMM
    }

    // final H is in Abuf[LAGS & 1] == Abuf[0]; reuse Abuf[1] as float scratch
    float* Ybuf = (float*)&Abuf[1][0];            // [64][64]
    float* O1f  = Ybuf + TILEB * Y_DIM;           // [64][64]

    // ---- head: Y = H@V + b ; O1 = tanh(Y@fc1 + b1) ; out = O1@fc2 + b2 ----
    for (int i = tid; i < TILEB * Y_DIM; i += 256) {
        int r = i >> 6, y = i & 63;
        float s = bias_out[y];
        for (int h = 0; h < HID; ++h)
            s += bf2f(Abuf[0][r * ROWSTRIDE + h]) * V[h * Y_DIM + y];
        Ybuf[i] = s;
    }
    __syncthreads();
    for (int i = tid; i < TILEB * Y_DIM; i += 256) {
        int r = i >> 6, y = i & 63;
        float s = fc1_b[y];
        for (int k = 0; k < Y_DIM; ++k)
            s += Ybuf[r * Y_DIM + k] * fc1_w[k * Y_DIM + y];
        O1f[i] = fast_tanh(s);
    }
    __syncthreads();
    if (tid < TILEB * 3) {
        int r = tid / 3, p = tid % 3;
        float s = fc2_b[p];
        for (int k = 0; k < Y_DIM; ++k)
            s += O1f[r * Y_DIM + k] * fc2_w[k * 3 + p];
        out[(size_t)(b0 + r) * 3 + p] = s;
    }
}

// ---------------------------------------------------------------------------
extern "C" void kernel_launch(void* const* d_in, const int* in_sizes, int n_in,
                              void* d_out, int out_size, void* d_ws, size_t ws_size,
                              hipStream_t stream)
{
    const float* X        = (const float*)d_in[0];
    const float* Wf       = (const float*)d_in[1];
    const float* Wi       = (const float*)d_in[2];
    const float* Wc       = (const float*)d_in[3];
    const float* Wo       = (const float*)d_in[4];
    const float* Uf       = (const float*)d_in[5];
    const float* Ui       = (const float*)d_in[6];
    const float* Uc       = (const float*)d_in[7];
    const float* Uo       = (const float*)d_in[8];
    const float* b_f      = (const float*)d_in[9];
    const float* b_i      = (const float*)d_in[10];
    const float* b_c      = (const float*)d_in[11];
    const float* b_o      = (const float*)d_in[12];
    const float* V        = (const float*)d_in[13];
    const float* bias_out = (const float*)d_in[14];
    const float* fc1_w    = (const float*)d_in[15];
    const float* fc1_b    = (const float*)d_in[16];
    const float* fc2_w    = (const float*)d_in[17];
    const float* fc2_b    = (const float*)d_in[18];

    unsigned short* packed = (unsigned short*)d_ws;   // 768 KB

    pack_weights<<<(KTOT * NTOT) / 256, 256, 0, stream>>>(
        Wf, Wi, Wc, Wo, Uf, Ui, Uc, Uo, packed);

    lstm_main<<<BATCH / TILEB, 256, 0, stream>>>(
        X, packed, b_f, b_i, b_c, b_o, V, bias_out,
        fc1_w, fc1_b, fc2_w, fc2_b, (float*)d_out);
}
